// _System_general_21406117003440
// MI455X (gfx1250) — matrix operand order repaired and checked by reading, not compiled, not run on hardware
//
#include <hip/hip_runtime.h>
#include <hip/hip_bf16.h>

// CDNA5 (gfx1250) tanh-triangular recurrent forward:
//   b  = xi@C1^T + u@D12^T
//   w[i] = tanh(b[i] + sum_{j<i} L[i,j] w[j]),  L = tril(D11,-1)   (blocked 16-wide)
//   out = xi@A^T + w@B1^T + u@B2^T
// Strategy: prelude kernels convert xi/u/weights to f16 in d_ws ONCE, so the
// main kernel's WMMA operands are pure 16B loads (no per-use v_cvt).
// Dense work: v_wmma_f32_16x16x32_f16. w tile lives in LDS (f16, 64KB).

typedef _Float16 v16h __attribute__((ext_vector_type(16)));
typedef _Float16 v8h  __attribute__((ext_vector_type(8)));
typedef _Float16 v4h  __attribute__((ext_vector_type(4)));
typedef float    v8f  __attribute__((ext_vector_type(8)));

#define NXD 256
#define NUD 64
#define NQD 512

static __device__ __forceinline__ v8f wmma_f16(v16h a, v16h b, v8f c) {
  // (neg_a, A, neg_b, B, c_mod, C, reuse_a, reuse_b)
  return __builtin_amdgcn_wmma_f32_16x16x32_f16(false, a, false, b, (short)0, c,
                                                false, false);
}

// ---------------- f32 -> f16 conversion prelude (4 elems/thread) -------------
__global__ void cvt_f32_to_f16_x4(const float* __restrict__ src,
                                  _Float16* __restrict__ dst, int n4) {
  int i = blockIdx.x * blockDim.x + threadIdx.x;
  if (i < n4) {
    const float4 q = ((const float4*)src)[i];
    v4h h;
    h[0] = (_Float16)q.x; h[1] = (_Float16)q.y;
    h[2] = (_Float16)q.z; h[3] = (_Float16)q.w;
    ((v4h*)dst)[i] = h;
  }
}

// A-operand fragment (16x32 f16) from a row-major f16 array (global or LDS).
// Lanes 0-15: K = k0+{0..7, 16..23}; lanes 16-31: K = k0+{8..15, 24..31}.
template <bool FULL>
static __device__ __forceinline__ v16h load_a16(const _Float16* __restrict__ base,
                                                int ld, int row, int k0) {
  const int hsel = (threadIdx.x >> 4) & 1;
  const _Float16* p = base + (size_t)row * ld + k0 + hsel * 8;
  v8h lo = *(const v8h*)p;                 // 16B load
  v8h hi = {};
  if (FULL) hi = *(const v8h*)(p + 16);
  return __builtin_shufflevector(lo, hi, 0, 1, 2, 3, 4, 5, 6, 7,
                                 8, 9, 10, 11, 12, 13, 14, 15);
}

// B-operand fragment (32x16 f16): lane n = output column n of W (row n of W^T).
// Lanes 0-15: K = k0+{0..7, 16..23}; lanes 16-31: K = k0+{8..15, 24..31}.
template <bool FULL>
static __device__ __forceinline__ v16h load_b16(const _Float16* __restrict__ base,
                                                int ld, int row, int k0) {
  const int hsel = (threadIdx.x >> 4) & 1;
  const _Float16* p = base + (size_t)row * ld + k0 + hsel * 8;
  v8h lo = *(const v8h*)p;
  v8h hi = {};  // K 16..31 invalid -> zeros
  if (FULL) hi = *(const v8h*)(p + 16);
  return __builtin_shufflevector(lo, hi, 0, 1, 2, 3, 4, 5, 6, 7,
                                 8, 9, 10, 11, 12, 13, 14, 15);
}

__global__ void __launch_bounds__(128)
ren_forward(const _Float16* __restrict__ xi16, const _Float16* __restrict__ u16,
            const _Float16* __restrict__ A16,  const _Float16* __restrict__ B116,
            const _Float16* __restrict__ B216, const _Float16* __restrict__ C116,
            const _Float16* __restrict__ D1116, const _Float16* __restrict__ D1216,
            const float* __restrict__ D11f, float* __restrict__ out) {
  __shared__ __align__(16) _Float16 wtile[64 * NQD];  // 64KB f16 w tile

  const int lane = threadIdx.x & 31;
  const int wv   = threadIdx.x >> 5;  // wave 0..3, owns 16 batch rows
  const int c    = lane & 15;         // column / A-row selector
  const int hsel = lane >> 4;
  const int lrow = wv * 16;                        // LDS row base of this wave
  const int rowA = blockIdx.x * 64 + lrow + c;     // global batch row (A frags)
  const int rowW = lrow + c;                       // LDS row (A frags)

  // Hoist xi/u A-fragments: loaded once, reused by all 32 blocks + out stage.
  v16h axi[8];
#pragma unroll
  for (int t = 0; t < 8; ++t) axi[t] = load_a16<true>(xi16, NXD, rowA, t * 32);
  v16h au[2];
#pragma unroll
  for (int t = 0; t < 2; ++t) au[t] = load_a16<true>(u16, NUD, rowA, t * 32);

  // ---- blocked triangular recurrence over 32 feature blocks of 16 ----
  for (int kb = 0; kb < 32; ++kb) {
    const int fb = kb * 16;
    v8f acc = {};

    // b-part: xi @ C1^T (K=256) + u @ D12^T (K=64)
#pragma unroll
    for (int t = 0; t < 8; ++t)
      acc = wmma_f16(axi[t], load_b16<true>(C116, NXD, fb + c, t * 32), acc);
#pragma unroll
    for (int t = 0; t < 2; ++t)
      acc = wmma_f16(au[t], load_b16<true>(D1216, NUD, fb + c, t * 32), acc);

    // coupling from previous blocks: w[:,0:fb] @ L[fb:fb+16, 0:fb]^T
    // (all entries strictly lower -> no masking needed)
    int kk = 0;
    for (; kk + 32 <= fb; kk += 32)
      acc = wmma_f16(load_a16<true>(wtile, NQD, rowW, kk),
                     load_b16<true>(D1116, NQD, fb + c, kk), acc);
    if (kk < fb)  // odd 16-wide remainder, upper K half zero-padded
      acc = wmma_f16(load_a16<false>(wtile, NQD, rowW, kk),
                     load_b16<false>(D1116, NQD, fb + c, kk), acc);

    // ---- serial 16-step solve of the strictly-lower 16x16 diagonal block ----
    // lane c owns column fb+c; VGPR r holds row (lrow + 8*hsel + r).  f32 math.
    float lr[16];
    const float* drow = D11f + (size_t)(fb + c) * NQD + fb;
#pragma unroll
    for (int i = 0; i < 16; ++i) lr[i] = (i < c) ? drow[i] : 0.f;

    float vcol[8];
#pragma unroll
    for (int r = 0; r < 8; ++r) vcol[r] = acc[r];

#pragma unroll
    for (int i = 0; i < 16; ++i) {
      if (c == i) {  // finalize column i
#pragma unroll
        for (int r = 0; r < 8; ++r) vcol[r] = tanhf(vcol[r]);
      }
      const int src = i + (hsel << 4);  // lane holding w_i for our row half
#pragma unroll
      for (int r = 0; r < 8; ++r)
        vcol[r] = fmaf(lr[i], __shfl(vcol[r], src, 32), vcol[r]);
    }

    // commit w block to LDS (f16), row-major for later ds_load_b128 A-frags
#pragma unroll
    for (int r = 0; r < 8; ++r)
      wtile[(size_t)(lrow + hsel * 8 + r) * NQD + fb + c] = (_Float16)vcol[r];
  }

  // ---- output: xi' = xi@A^T + w@B1^T + u@B2^T, [16 x 256] per wave ----
  for (int nt = 0; nt < 16; ++nt) {
    const int nb = nt * 16;
    v8f acc = {};
#pragma unroll
    for (int t = 0; t < 8; ++t)
      acc = wmma_f16(axi[t], load_b16<true>(A16, NXD, nb + c, t * 32), acc);
    for (int kk = 0; kk < NQD; kk += 32)
      acc = wmma_f16(load_a16<true>(wtile, NQD, rowW, kk),
                     load_b16<true>(B116, NQD, nb + c, kk), acc);
#pragma unroll
    for (int t = 0; t < 2; ++t)
      acc = wmma_f16(au[t], load_b16<true>(B216, NUD, nb + c, t * 32), acc);

    const int ro = blockIdx.x * 64 + lrow + hsel * 8;
#pragma unroll
    for (int r = 0; r < 8; ++r)
      out[(size_t)(ro + r) * NXD + nb + c] = acc[r];
  }
}

extern "C" void kernel_launch(void* const* d_in, const int* in_sizes, int n_in,
                              void* d_out, int out_size, void* d_ws, size_t ws_size,
                              hipStream_t stream) {
  // setup_inputs order: t, xi, u, A, B1, B2, C1, D11, D12
  const float* xi  = (const float*)d_in[1];
  const float* u   = (const float*)d_in[2];
  const float* A   = (const float*)d_in[3];
  const float* B1  = (const float*)d_in[4];
  const float* B2  = (const float*)d_in[5];
  const float* C1  = (const float*)d_in[6];
  const float* D11 = (const float*)d_in[7];
  const float* D12 = (const float*)d_in[8];
  float* out = (float*)d_out;

  const int N = in_sizes[1] / NXD;  // 65536

  // f16 workspace layout (element counts; all multiples of 8 -> 16B aligned)
  _Float16* w = (_Float16*)d_ws;
  _Float16* xi16  = w;                 w += (size_t)N * NXD;
  _Float16* u16   = w;                 w += (size_t)N * NUD;
  _Float16* A16   = w;                 w += NXD * NXD;
  _Float16* B116  = w;                 w += NXD * NQD;
  _Float16* B216  = w;                 w += NXD * NUD;
  _Float16* C116  = w;                 w += NQD * NXD;
  _Float16* D1116 = w;                 w += NQD * NQD;
  _Float16* D1216 = w;                 /* end */

  // Prelude: one-time f32 -> f16 conversion of activations + weights.
  const int T = 256;
  auto cvt = [&](const float* s, _Float16* d, size_t n) {
    int n4 = (int)(n / 4);
    cvt_f32_to_f16_x4<<<(n4 + T - 1) / T, T, 0, stream>>>(s, d, n4);
  };
  cvt(xi,  xi16,  (size_t)N * NXD);
  cvt(u,   u16,   (size_t)N * NUD);
  cvt(A,   A16,   (size_t)NXD * NXD);
  cvt(B1,  B116,  (size_t)NXD * NQD);
  cvt(B2,  B216,  (size_t)NXD * NUD);
  cvt(C1,  C116,  (size_t)NQD * NXD);
  cvt(D11, D1116, (size_t)NQD * NQD);
  cvt(D12, D1216, (size_t)NQD * NUD);

  dim3 grid(N / 64), block(128);  // 4 waves/WG, 16 batch rows per wave
  ren_forward<<<grid, block, 0, stream>>>(xi16, u16, A16, B116, B216, C116,
                                          D1116, D1216, D11, out);
}
